// Model_52183852646989
// MI455X (gfx1250) — matrix operand order repaired and checked by reading, not compiled, not run on hardware
//
#include <hip/hip_runtime.h>
#include <hip/hip_bf16.h>

// ---------------- WMMA types & helpers (gfx1250, wave32) ----------------
typedef __attribute__((ext_vector_type(16))) __bf16 v16bf;
typedef __attribute__((ext_vector_type(8)))  __bf16 v8bf;
typedef __attribute__((ext_vector_type(8)))  float  v8f;

__device__ __forceinline__ v8f wmma_bf16(v16bf a, v16bf b, v8f c) {
  return __builtin_amdgcn_wmma_f32_16x16x32_bf16(false, a, false, b, (short)0, c,
                                                 false, false);
}

__device__ __forceinline__ v16bf join16(v8bf lo, v8bf hi) {
  return __builtin_shufflevector(lo, hi, 0, 1, 2, 3, 4, 5, 6, 7,
                                 8, 9, 10, 11, 12, 13, 14, 15);
}

// A fragment built from fp32 weights on the fly (used by small k1 kernels only).
// lane L: row m = L&15; K-base 0/16 per lane half, contiguous k = 16h + s.
__device__ __forceinline__ v16bf load_afrag_w(const float* __restrict__ w, int Cin,
                                              int ksize, int oc0, int ic0, int kk,
                                              int lane) {
  int m  = lane & 15;
  int kb = (lane & 16) ? 16 : 0;
  const float* wr = w + ((size_t)(oc0 + m) * Cin) * ksize + kk;
  v16bf a;
#pragma unroll
  for (int e = 0; e < 8; ++e) {
    a[e]     = (__bf16)wr[(size_t)(ic0 + kb + e) * ksize];
    a[e + 8] = (__bf16)wr[(size_t)(ic0 + kb + 8 + e) * ksize];
  }
  return a;
}

// B fragment straight from global fp32 activations (k_upctxt only).
__device__ __forceinline__ v16bf load_bfrag_act(const float* __restrict__ actRow,
                                                int storeT, int Text, int upshift,
                                                int ic0, int tbase, int lane) {
  int n  = lane & 15;
  int kb = (lane & 16) ? 16 : 0;
  int u  = tbase + n;
  bool ok = (u >= 0) && (u < Text);
  int idx = ok ? (u >> upshift) : 0;
  const float* ap = actRow + idx;
  v16bf b;
#pragma unroll
  for (int e = 0; e < 16; ++e) {
    float v = ok ? ap[(size_t)(ic0 + kb + e) * storeT] : 0.0f;
    b[e] = (__bf16)v;
  }
  return b;
}

// ---------------- weight pre-pack: fp32 [Cout][128][k] -> bf16 A-fragment order ----
// frag f = (octile*ksize + kk)*4 + icb ; within frag: lane*16 contiguous halves.
// Rows with oc >= cout_real are zero-filled (no OOB reads, padded oc-tiles valid).
__global__ void k_packw(const float* __restrict__ w, __bf16* __restrict__ wp,
                        int ntiles, int cout_real, int ksize) {
  int idx = blockIdx.x * blockDim.x + threadIdx.x;
  int total = ntiles * ksize * 4 * 32;
  if (idx >= total) return;
  int lane = idx & 31;
  int f = idx >> 5;
  int icb = f & 3;
  int kk = (f >> 2) % ksize;
  int ot = f / (ksize * 4);
  int m = lane & 15;
  int kb = (lane & 16) ? 16 : 0;
  int oc = ot * 16 + m;
  __bf16 h[16];
#pragma unroll
  for (int e = 0; e < 8; ++e) {
    int ic0 = icb * 32 + kb + e;
    int ic1 = icb * 32 + kb + 8 + e;
    h[e]     = (oc < cout_real) ? (__bf16)w[((size_t)oc * 128 + ic0) * ksize + kk]
                                : (__bf16)0.f;
    h[e + 8] = (oc < cout_real) ? (__bf16)w[((size_t)oc * 128 + ic1) * ksize + kk]
                                : (__bf16)0.f;
  }
  __bf16* dst = wp + ((size_t)f << 9) + (lane << 4);
#pragma unroll
  for (int e = 0; e < 16; ++e) dst[e] = h[e];
}

// ---------------- implicit-GEMM conv1d, Cin = 128, LDS-staged B ----------------
// grid: x = ceil(Tout/64), y = rows; block = (Cout/16) waves, one oc-tile per wave.
// Per tap kk: stage bf16 strip [t 0..63][ch 0..127] (transposed, padded stride 136)
// so a B fragment = 16 contiguous halves = two ds_load_b128.  A fragments are two
// aligned global b128 loads from the pre-packed bf16 weights.  4 WMMAs share each A.
#define STRIDE 136
__global__ void k_conv_wmma(const float* __restrict__ act,
                            const __bf16* __restrict__ wp,
                            const float* __restrict__ bias,
                            const float* __restrict__ resid, float* __restrict__ out,
                            int Tout, int Text, int storeT, int upshift,
                            int ksize, int dil, int pad, int do_leaky, int cout_real) {
  __shared__ __bf16 smem[64 * STRIDE];
  int tid = threadIdx.x;
  int lane = tid & 31;
  int octile = tid >> 5;
  int t0 = blockIdx.x * 64;
  int row = blockIdx.y;
  const float* actRow = act + (size_t)row * 128 * storeT;
  if (t0 + 64 < storeT) __builtin_prefetch(actRow + t0 + 64, 0, 1);
  v8f acc[4] = {};
  int n = lane & 15;
  int kb2 = (lane & 16) ? 16 : 0;
  for (int kk = 0; kk < ksize; ++kk) {
    int off = kk * dil - pad;
    __syncthreads();
    for (int e2 = tid; e2 < 128 * 64; e2 += blockDim.x) {
      int tl = e2 & 63;
      int ch = e2 >> 6;
      int u = t0 + off + tl;
      float v = (u >= 0 && u < Text)
                    ? actRow[(size_t)ch * storeT + (u >> upshift)] : 0.0f;
      smem[tl * STRIDE + ch] = (__bf16)v;
    }
    __syncthreads();
#pragma unroll
    for (int icb = 0; icb < 4; ++icb) {
      const __bf16* ap = wp + (((size_t)(octile * ksize + kk) * 4 + icb) << 9)
                            + (lane << 4);
      v16bf a = join16(*(const v8bf*)ap, *(const v8bf*)(ap + 8));
#pragma unroll
      for (int j = 0; j < 4; ++j) {
        const __bf16* bp = &smem[(n + j * 16) * STRIDE + icb * 32 + kb2];
        v16bf b = join16(*(const v8bf*)bp, *(const v8bf*)(bp + 8));
        acc[j] = wmma_bf16(a, b, acc[j]);
      }
    }
  }
  int mo = (lane & 16) ? 8 : 0;
#pragma unroll
  for (int r = 0; r < 8; ++r) {
    int oc = octile * 16 + mo + r;
    if (oc >= cout_real) continue;
    float bv = bias[oc];
    size_t base = ((size_t)row * cout_real + oc) * Tout;
#pragma unroll
    for (int j = 0; j < 4; ++j) {
      int t = t0 + j * 16 + n;
      if (t < Tout) {
        float v = acc[j][r] + bv;
        if (resid) v += resid[base + t];
        if (do_leaky) v = (v >= 0.f) ? v : 0.2f * v;
        out[base + t] = v;
      }
    }
  }
}

// ---------------- Filterbank conv: Cin=1, 512 taps as GEMM K; x staged in LDS ----
__global__ void k_fbank(const float* __restrict__ x, const float* __restrict__ fw,
                        const float* __restrict__ fb, float* __restrict__ out, int T) {
  __shared__ __bf16 xs[544];
  int tid = threadIdx.x;
  int lane = tid & 31;
  int b0 = (tid >> 5) << 4;
  int t0 = blockIdx.x * 32;
  int row = blockIdx.y;
  const float* xr = x + (size_t)row * T;
  for (int j = tid; j < 544; j += blockDim.x) {
    int u = t0 - 256 + j;
    xs[j] = (__bf16)((u >= 0 && u < T) ? xr[u] : 0.0f);
  }
  __syncthreads();
  int n = lane & 15;
  int kb = (lane & 16) ? 16 : 0;
  v8f acc0 = {}; v8f acc1 = {};
  for (int k0 = 0; k0 < 512; k0 += 32) {
    v16bf a = load_afrag_w(fw, 512, 1, b0, k0, 0, lane);
    v16bf bb0, bb1;
    int base = n + k0 + kb;
#pragma unroll
    for (int e = 0; e < 16; ++e) {
      bb0[e] = xs[base + e];
      bb1[e] = xs[base + 16 + e];
    }
    acc0 = wmma_bf16(a, bb0, acc0);
    acc1 = wmma_bf16(a, bb1, acc1);
  }
  int mo = (lane & 16) ? 8 : 0;
#pragma unroll
  for (int r = 0; r < 8; ++r) {
    int band = b0 + mo + r;
    float bv = fb[band];
    size_t base = ((size_t)row * 128 + band) * T;
    out[base + t0 + n]      = acc0[r] + bv;
    out[base + t0 + 16 + n] = acc1[r] + bv;
  }
}

// ------------- up-projection (k1, 128->16) + ReLU + sum over T into ctxt ----------
__global__ void k_upctxt(const float* __restrict__ act, const float* __restrict__ w,
                         const float* __restrict__ bias, float* __restrict__ ctxt,
                         int T) {
  __shared__ float ls[16];
  int lane = threadIdx.x;  // blockDim = 32 (one wave)
  if (lane < 16) ls[lane] = 0.f;
  __syncthreads();
  int t0  = blockIdx.x * 32;
  int row = blockIdx.y;
  const float* actRow = act + (size_t)row * 128 * T;
  v8f acc0 = {}; v8f acc1 = {};
#pragma unroll
  for (int ic0 = 0; ic0 < 128; ic0 += 32) {
    v16bf a  = load_afrag_w(w, 128, 1, 0, ic0, 0, lane);
    v16bf b0 = load_bfrag_act(actRow, T, T, 0, ic0, t0, lane);
    acc0 = wmma_bf16(a, b0, acc0);
    v16bf b1 = load_bfrag_act(actRow, T, T, 0, ic0, t0 + 16, lane);
    acc1 = wmma_bf16(a, b1, acc1);
  }
  int mo = (lane & 16) ? 8 : 0;
#pragma unroll
  for (int r = 0; r < 8; ++r) {
    int oc = mo + r;
    float bv = bias[oc];
    float v0 = fmaxf(acc0[r] + bv, 0.f);
    float v1 = fmaxf(acc1[r] + bv, 0.f);
    atomicAdd(&ls[oc], v0 + v1);
  }
  __syncthreads();
  if (lane < 16) atomicAdd(&ctxt[row * 16 + lane], ls[lane]);
}

// ---------------- BatchNorm: stats (atomic) + apply ----------------
__global__ void k_zero(float* __restrict__ p, int n) {
  for (int i = blockIdx.x * blockDim.x + threadIdx.x; i < n; i += gridDim.x * blockDim.x)
    p[i] = 0.f;
}

__global__ void k_bnstats(const float* __restrict__ y, float* __restrict__ stats,
                          int C, int rows, int T) {
  int c = blockIdx.y;
  long total = (long)rows * T;
  float s = 0.f, s2 = 0.f;
  for (long i = (long)blockIdx.x * blockDim.x + threadIdx.x; i < total;
       i += (long)gridDim.x * blockDim.x) {
    long r = i / T;
    long t = i - r * T;
    float v = y[((size_t)r * C + c) * T + t];
    s += v; s2 += v * v;
  }
  __shared__ float sa[256], sb[256];
  int tid = threadIdx.x;
  sa[tid] = s; sb[tid] = s2; __syncthreads();
  for (int o = 128; o > 0; o >>= 1) {
    if (tid < o) { sa[tid] += sa[tid + o]; sb[tid] += sb[tid + o]; }
    __syncthreads();
  }
  if (tid == 0) { atomicAdd(&stats[c], sa[0]); atomicAdd(&stats[C + c], sb[0]); }
}

__global__ void k_bnapply(float* __restrict__ y, const float* __restrict__ stats,
                          const float* __restrict__ g, const float* __restrict__ b,
                          int C, int T, long total, float invN) {
  long i = (long)blockIdx.x * blockDim.x + threadIdx.x;
  if (i >= total) return;
  int c = (int)((i / T) % C);
  float mean = stats[c] * invN;
  float var  = stats[C + c] * invN - mean * mean;
  float v = y[i];
  y[i] = (v - mean) * rsqrtf(var + 1e-5f) * g[c] + b[c];
}

// ---------------- LinearOutputStack (in->128, 3x[lin,LN,leaky], 128->O) ----------
__global__ void k_los(const float* __restrict__ in, int IN, int O,
                      const float* wi, const float* bi,
                      const float* w0, const float* b0l, const float* g0, const float* bb0,
                      const float* w1, const float* b1l, const float* g1, const float* bb1,
                      const float* w2, const float* b2l, const float* g2, const float* bb2,
                      const float* wo, const float* bo, float* __restrict__ outp) {
  int r = blockIdx.x;
  int j = threadIdx.x;  // 128 threads
  __shared__ float xin[128], h[128], red[128];
  if (j < IN) xin[j] = in[r * IN + j];
  __syncthreads();
  float a = bi[j];
  for (int i = 0; i < IN; ++i) a += xin[i] * wi[i * 128 + j];
  h[j] = a;
  const float* W[3]  = {w0, w1, w2};
  const float* Bb[3] = {b0l, b1l, b2l};
  const float* G[3]  = {g0, g1, g2};
  const float* BB[3] = {bb0, bb1, bb2};
  for (int l = 0; l < 3; ++l) {
    __syncthreads();
    float p = Bb[l][j];
    for (int i = 0; i < 128; ++i) p += h[i] * W[l][i * 128 + j];
    __syncthreads();
    red[j] = p; __syncthreads();
    for (int o = 64; o > 0; o >>= 1) { if (j < o) red[j] += red[j + o]; __syncthreads(); }
    float mean = red[0] * (1.0f / 128.0f);
    __syncthreads();
    float d = p - mean;
    red[j] = d * d; __syncthreads();
    for (int o = 64; o > 0; o >>= 1) { if (j < o) red[j] += red[j + o]; __syncthreads(); }
    float var = red[0] * (1.0f / 128.0f);
    __syncthreads();
    float v = d * rsqrtf(var + 1e-5f) * G[l][j] + BB[l][j];
    h[j] = (v >= 0.f) ? v : 0.2f * v;
  }
  __syncthreads();
  if (j < O) {
    float ov = bo[j];
    for (int i = 0; i < 128; ++i) ov += h[i] * wo[i * O + j];
    outp[r * O + j] = ov;
  }
}

// z[b*16+e][l] = los_out[b][l] + latent[e][l]
__global__ void k_addlat(const float* __restrict__ lo, const float* __restrict__ lat,
                         float* __restrict__ z) {
  int idx = blockIdx.x * blockDim.x + threadIdx.x;  // 4*16*16
  if (idx >= 1024) return;
  int l = idx & 15, e = (idx >> 4) & 15, b = idx >> 8;
  z[idx] = lo[b * 16 + l] + lat[e * 16 + l];
}

// small dense: out[r][o] = b[o] + sum_i in[r][i]*w[i][o]; block = O threads
__global__ void k_linear(const float* __restrict__ in, const float* __restrict__ w,
                         const float* __restrict__ b, float* __restrict__ outp,
                         int IN, int O) {
  int r = blockIdx.x, o = threadIdx.x;
  float a = b[o];
  for (int i = 0; i < IN; ++i) a += in[r * IN + i] * w[i * O + o];
  outp[(size_t)r * O + o] = a;
}

// |frames| + torch-style linear interp 16 -> 4096
__global__ void k_interp(const float* __restrict__ fr, float* __restrict__ outp) {
  int idx = blockIdx.x * blockDim.x + threadIdx.x;  // 64*4096
  int r = idx >> 12, t = idx & 4095;
  float pos = (t + 0.5f) * (16.0f / 4096.0f) - 0.5f;
  pos = fminf(fmaxf(pos, 0.0f), 15.0f);
  int i0 = (int)floorf(pos);
  int i1 = (i0 + 1 > 15) ? 15 : i0 + 1;
  float wt = pos - (float)i0;
  float f0 = fabsf(fr[r * 16 + i0]);
  float f1 = fabsf(fr[r * 16 + i1]);
  outp[idx] = f0 * (1.0f - wt) + f1 * wt;
}

// filt[r][k] = sum_f fmix[r][f] * mfilters[f][k]
__global__ void k_makefilt(const float* __restrict__ fmix, const float* __restrict__ mf,
                           float* __restrict__ filt) {
  int idx = blockIdx.x * blockDim.x + threadIdx.x;  // 64*16
  if (idx >= 1024) return;
  int r = idx >> 4, kk = idx & 15;
  float s = 0.f;
  for (int f = 0; f < 8; ++f) s += fmix[r * 8 + f] * mf[f * 16 + kk];
  filt[idx] = s;
}

// impulses = (noise (*) 16-tap filt) * frames, zero-padded 4096 -> 16384
__global__ void k_fir(const float* __restrict__ noise, const float* __restrict__ filt,
                      const float* __restrict__ frames, float* __restrict__ imps) {
  int idx = blockIdx.x * blockDim.x + threadIdx.x;  // 64*16384
  int r = idx >> 14, t = idx & 16383;
  float v = 0.f;
  if (t < 4096) {
    float s = 0.f;
#pragma unroll
    for (int tau = 0; tau < 16; ++tau)
      if (tau <= t) s += filt[r * 16 + tau] * noise[r * 4096 + (t - tau)];
    v = s * frames[r * 4096 + t];
  }
  imps[idx] = v;
}

// out = imps*softmax0 + (imps (*) res)*softmax1 ; Toeplitz correlation, LDS-chunked
__global__ void k_final(const float* __restrict__ imp, const float* __restrict__ res,
                        const float* __restrict__ mixraw, float* __restrict__ out) {
  int r = blockIdx.y;
  int t = blockIdx.x * 256 + threadIdx.x;
  __shared__ float la[256];
  const float* impR = imp + (size_t)r * 16384;
  const float* resR = res + (size_t)r * 16384;
  float acc = 0.f;
  for (int c0 = 0; c0 < 4096; c0 += 256) {
    __syncthreads();
    la[threadIdx.x] = impR[c0 + threadIdx.x];
    __syncthreads();
    for (int j = 0; j < 256; ++j) {
      int tau = c0 + j;
      if (tau > t) break;
      acc += la[j] * resR[t - tau];
    }
  }
  float m0 = mixraw[r * 2], m1 = mixraw[r * 2 + 1];
  float mx = fmaxf(m0, m1);
  float e0 = expf(m0 - mx), e1 = expf(m1 - mx);
  float inv = 1.0f / (e0 + e1);
  out[(size_t)r * 16384 + t] = impR[t] * (e0 * inv) + acc * (e1 * inv);
}

// =============================== host driver ===============================
extern "C" void kernel_launch(void* const* d_in, const int* in_sizes, int n_in,
                              void* d_out, int out_size, void* d_ws, size_t ws_size,
                              hipStream_t stream) {
  (void)in_sizes; (void)n_in; (void)out_size; (void)ws_size;
  const int T = 32768;
  int p = 0;
  auto NEXT = [&]() { return (const float*)d_in[p++]; };

  const float* x     = NEXT();
  const float* noise = NEXT();
  const float* fb_w  = NEXT();
  const float* fb_b  = NEXT();
  const float* stem_w = NEXT();
  const float* stem_b = NEXT();
  const float* stem_g = NEXT();
  const float* stem_bb = NEXT();
  struct Blk { const float *wd, *bd, *wc, *bc, *g, *bb; } blk[7];
  for (int i = 0; i < 7; ++i) {
    blk[i].wd = NEXT(); blk[i].bd = NEXT(); blk[i].wc = NEXT();
    blk[i].bc = NEXT(); blk[i].g  = NEXT(); blk[i].bb = NEXT();
  }
  const float* up_w = NEXT();
  const float* up_b = NEXT();
  const float *EI[16], *ER[16], *EM[16], *FM[16], *ML[16];
  for (int i = 0; i < 16; ++i) EI[i] = NEXT();
  for (int i = 0; i < 16; ++i) ER[i] = NEXT();
  for (int i = 0; i < 16; ++i) EM[i] = NEXT();
  const float* imp_latent = NEXT();
  const float* res_latent = NEXT();
  const float* mix_latent = NEXT();
  const float* ie_pw = NEXT(); const float* ie_pb = NEXT();
  const float* ie_s0w = NEXT(); const float* ie_s0b = NEXT();
  const float* ie_s0g = NEXT(); const float* ie_s0bb = NEXT();
  const float* ie_s1w = NEXT(); const float* ie_s1b = NEXT();
  for (int i = 0; i < 16; ++i) FM[i] = NEXT();
  const float* mfilters = NEXT();
  const float* ru_pw = NEXT(); const float* ru_pb = NEXT();
  const float *rs_w[11], *rs_b[11], *rs_g[10], *rs_bb[10];
  for (int s = 0; s < 10; ++s) {
    rs_w[s] = NEXT(); rs_b[s] = NEXT(); rs_g[s] = NEXT(); rs_bb[s] = NEXT();
  }
  rs_w[10] = NEXT(); rs_b[10] = NEXT();
  for (int i = 0; i < 16; ++i) ML[i] = NEXT();

  // workspace layout (floats)
  float* ws = (float*)d_ws;
  size_t ofs = 0;
  auto alloc = [&](size_t n) { float* q = ws + ofs; ofs += (n + 7) & ~(size_t)7; return q; };
  float* bufA = alloc((size_t)4 * 128 * T);
  float* bufB = alloc((size_t)4 * 128 * T);
  float* bufT = alloc((size_t)4 * 128 * T);
  float* stats = alloc(256);
  float* ctxt  = alloc(64);
  float* losout = alloc(64);
  float* zimp = alloc(1024);
  float* zres = alloc(1024);
  float* zmix = alloc(1024);
  float* pimp = alloc(64 * 512);
  float* e0buf = alloc((size_t)64 * 128 * 8);
  float* framesraw = alloc(64 * 16);
  float* frames = alloc((size_t)64 * 4096);
  float* fmixb = alloc(64 * 8);
  float* filt  = alloc(64 * 16);
  float* imps  = alloc((size_t)64 * 16384);
  float* pres  = alloc(64 * 1024);
  float* r0    = alloc((size_t)64 * 128 * 4096);
  float* r1    = alloc((size_t)64 * 128 * 8192);
  float* resbuf = alloc((size_t)64 * 16384);
  float* mixout = alloc(128);
  float* packRegion = alloc(600000);  // bf16 fragment-packed weights (~2.3 MB)

  // ---- pre-pack all 128-channel conv weights to bf16 fragment layout ----
  __bf16* packBase = (__bf16*)packRegion;
  size_t pofs = 0;
  auto pack = [&](const float* w, int ntiles, int cr, int ks) {
    __bf16* dst = packBase + pofs;
    int total = ntiles * ks * 4 * 32;
    k_packw<<<(total + 255) / 256, 256, 0, stream>>>(w, dst, ntiles, cr, ks);
    pofs += (size_t)ntiles * ks * 4 * 512;
    return (const __bf16*)dst;
  };
  const __bf16* P_stem = pack(stem_w, 8, 128, 7);
  const __bf16 *P_bd[7], *P_bc[7];
  for (int i = 0; i < 7; ++i) {
    P_bd[i] = pack(blk[i].wd, 8, 128, 3);
    P_bc[i] = pack(blk[i].wc, 8, 128, 1);
  }
  const __bf16* P_is0 = pack(ie_s0w, 8, 128, 3);
  const __bf16* P_is1 = pack(ie_s1w, 1, 1, 3);
  const __bf16* P_rs[11];
  for (int s = 0; s < 10; ++s) P_rs[s] = pack(rs_w[s], 8, 128, 3);
  P_rs[10] = pack(rs_w[10], 1, 1, 3);

  auto conv = [&](const float* a, const __bf16* wp, const float* bs, const float* rsd,
                  float* outp, int rows, int Cout, int Tout, int Text,
                  int storeT, int ups, int ks, int dl, int pd, int lk, int cr) {
    dim3 g((Tout + 63) / 64, rows);
    int thr = (Cout / 16) * 32;
    k_conv_wmma<<<g, thr, 0, stream>>>(a, wp, bs, rsd, outp, Tout, Text,
                                       storeT, ups, ks, dl, pd, lk, cr);
  };
  auto bn = [&](float* y, const float* g_, const float* b_, int C, int rows, int Tl) {
    k_zero<<<1, 256, 0, stream>>>(stats, 2 * C);
    long tot = (long)rows * Tl;
    int gx = (int)((tot + 255) / 256); if (gx > 256) gx = 256; if (gx < 1) gx = 1;
    k_bnstats<<<dim3(gx, C), 256, 0, stream>>>(y, stats, C, rows, Tl);
    long n = (long)rows * C * Tl;
    k_bnapply<<<(unsigned)((n + 255) / 256), 256, 0, stream>>>(
        y, stats, g_, b_, C, Tl, n, 1.0f / (float)tot);
  };
  auto los = [&](const float* in, int O, const float* const* P, float* outp, int rows) {
    k_los<<<rows, 128, 0, stream>>>(in, 16, O, P[0], P[1], P[2], P[3], P[4], P[5],
                                    P[6], P[7], P[8], P[9], P[10], P[11], P[12],
                                    P[13], P[14], P[15], outp);
  };

  // ---- main path: fbank -> stem -> 7 dilated blocks -> up+ctxt ----
  k_fbank<<<dim3(T / 32, 4), 256, 0, stream>>>(x, fb_w, fb_b, bufA, T);
  conv(bufA, P_stem, stem_b, nullptr, bufB, 4, 128, T, T, T, 0, 7, 1, 3, 1, 128);
  bn(bufB, stem_g, stem_bb, 128, 4, T);
  float* cur = bufB; float* nxt = bufA;
  const int dils[7] = {1, 3, 9, 27, 81, 243, 1};
  for (int i = 0; i < 7; ++i) {
    conv(cur, P_bd[i], blk[i].bd, nullptr, bufT, 4, 128, T, T, T, 0,
         3, dils[i], dils[i], 0, 128);
    conv(bufT, P_bc[i], blk[i].bc, cur, nxt, 4, 128, T, T, T, 0,
         1, 1, 0, 1, 128);
    bn(nxt, blk[i].g, blk[i].bb, 128, 4, T);
    float* t_ = cur; cur = nxt; nxt = t_;
  }
  k_zero<<<1, 64, 0, stream>>>(ctxt, 64);
  k_upctxt<<<dim3(T / 32, 4), 32, 0, stream>>>(cur, up_w, up_b, ctxt, T);

  // ---- latent embeddings ----
  los(ctxt, 16, EI, losout, 4);
  k_addlat<<<4, 256, 0, stream>>>(losout, imp_latent, zimp);
  los(ctxt, 16, ER, losout, 4);
  k_addlat<<<4, 256, 0, stream>>>(losout, res_latent, zres);
  los(ctxt, 16, EM, losout, 4);
  k_addlat<<<4, 256, 0, stream>>>(losout, mix_latent, zmix);

  // ---- GenerateImpulse ----
  k_linear<<<64, 512, 0, stream>>>(zimp, ie_pw, ie_pb, pimp, 16, 512);
  conv(pimp, P_is0, ie_s0b, nullptr, e0buf, 64, 128, 8, 8, 4, 1, 3, 1, 1, 1, 128);
  bn(e0buf, ie_s0g, ie_s0bb, 128, 64, 8);
  conv(e0buf, P_is1, ie_s1b, nullptr, framesraw, 64, 16, 16, 16, 8, 1, 3, 1, 1, 0, 1);
  k_interp<<<(64 * 4096) / 256, 256, 0, stream>>>(framesraw, frames);
  los(zimp, 8, FM, fmixb, 64);
  k_makefilt<<<4, 256, 0, stream>>>(fmixb, mfilters, filt);
  k_fir<<<(64 * 16384) / 256, 256, 0, stream>>>(noise, filt, frames, imps);

  // ---- resonance ConvUpsample: 11 stages, upsample folded into conv ----
  k_linear<<<64, 1024, 0, stream>>>(zres, ru_pw, ru_pb, pres, 16, 1024);
  float* rin = pres; int Tin = 8;
  for (int s = 0; s < 11; ++s) {
    int To = Tin * 2;
    bool last = (s == 10);
    float* ro = last ? resbuf : ((s & 1) ? r1 : r0);
    conv(rin, P_rs[s], rs_b[s], nullptr, ro, 64, last ? 16 : 128, To, To, Tin,
         1, 3, 1, 1, last ? 0 : 1, last ? 1 : 128);
    if (!last) bn(ro, rs_g[s], rs_bb[s], 128, 64, To);
    rin = ro; Tin = To;
  }

  // ---- mix + final combine ----
  los(zmix, 2, ML, mixout, 64);
  k_final<<<dim3(64, 64), 256, 0, stream>>>(imps, resbuf, mixout, (float*)d_out);
}
